// VQCodebook_10204842295874
// MI455X (gfx1250) — matrix operand order repaired and checked by reading, not compiled, not run on hardware
//
#include <hip/hip_runtime.h>
#include <hip/hip_bf16.h>

#define N_PTS     32768
#define NUM_CODES 8192
#define DIM       512

#define WAVES   8
#define MTILE   (16 * WAVES)   // 128 rows per block
#define CTILE   32             // codes per LDS tile
#define LDS_STR 520            // 512 bf16 + 8 pad -> 1040B row stride (16B aligned)
#define NTILES  (NUM_CODES / CTILE)

typedef __bf16 v16bf __attribute__((ext_vector_type(16)));
typedef float  v8f   __attribute__((ext_vector_type(8)));
typedef unsigned int       u32;
typedef unsigned long long u64;
typedef u32 u32x4 __attribute__((ext_vector_type(4)));
typedef u32 u32x8 __attribute__((ext_vector_type(8)));

__device__ __forceinline__ __bf16 cvt_bf16(float f) {
    union { float f; unsigned int u; } in; in.f = f;
    unsigned int u = in.u;
    u += 0x7FFFu + ((u >> 16) & 1u);            // round-to-nearest-even
    union { unsigned short s; __bf16 h; } o;
    o.s = (unsigned short)(u >> 16);
    return o.h;
}

// LDS byte offset of a __shared__ pointer (generic -> addrspace(3) cast)
__device__ __forceinline__ u32 lds_addr_of(const void* p) {
    return (u32)(u64)(__attribute__((address_space(3))) const void*)p;
}

// TDM: DMA one emb tile (CTILE rows x 512 bf16, 16B pad per 1024B row) into LDS.
// D# per CDNA5 ISA. Issue from ONE wave (uniform branch);
// completion tracked on TENSORcnt.
__device__ __forceinline__ void tdm_load_tile(const unsigned short* gsrc, u32 lds_off) {
    u64 ga = (u64)(const void*)gsrc;
    u32x4 g0;
    g0[0] = 0x1u;                                        // count=1, user mode
    g0[1] = lds_off;                                     // lds_addr
    g0[2] = (u32)(ga & 0xFFFFFFFFu);                     // global_addr[31:0]
    g0[3] = (u32)((ga >> 32) & 0x01FFFFFFu) | (2u << 30);// global_addr[56:32] | type=2
    u32x8 g1;
    g1[0] = (1u << 16)      // data_size = 2 bytes
          | (1u << 20)      // pad_enable
          | (7u << 22)      // pad_interval: 256 DWORDs = 1024 B
          | (3u << 25);     // pad_amount: 4 DWORDs = 16 B
    g1[1] = ((u32)DIM) << 16;                            // tensor_dim0[15:0] = 512
    g1[2] = ((u32)NUM_CODES) << 16;                      // dim0 hi=0 | tensor_dim1 lo = 8192
    g1[3] = ((u32)DIM) << 16;                            // dim1 hi=0 | tile_dim0 = 512
    g1[4] = (u32)CTILE;                                  // tile_dim1 = 32 | tile_dim2 = 0
    g1[5] = (u32)DIM;                                    // tensor_dim0_stride = 512
    g1[6] = 0u;                                          // stride0 hi | dim1_stride lo
    g1[7] = 0u;
    asm volatile("tensor_load_to_lds %0, %1" :: "s"(g0), "s"(g1) : "memory");
}

// ---------------------------------------------------------------------------
// Kernel 1: emb f32 -> bf16 (ws) and per-code squared norms (f32)
// ---------------------------------------------------------------------------
__global__ __launch_bounds__(256) void vq_prep(const float* __restrict__ emb,
                                               unsigned short* __restrict__ embBf,
                                               float* __restrict__ norms) {
    const int c = blockIdx.x;
    const float* row = emb + (size_t)c * DIM;
    unsigned short* orow = embBf + (size_t)c * DIM;
    float s = 0.f;
    for (int e = threadIdx.x; e < DIM; e += 256) {
        float v = row[e];
        s += v * v;
        union { __bf16 h; unsigned short u; } cv; cv.h = cvt_bf16(v);
        orow[e] = cv.u;
    }
    __shared__ float red[256];
    red[threadIdx.x] = s;
    __syncthreads();
    for (int off = 128; off > 0; off >>= 1) {
        if (threadIdx.x < off) red[threadIdx.x] += red[threadIdx.x + off];
        __syncthreads();
    }
    if (threadIdx.x == 0) norms[c] = red[0];
}

// ---------------------------------------------------------------------------
// Kernel 2: fused GEMM (bf16 WMMA, f32 accum) + argmin, TDM double-buffered
// ---------------------------------------------------------------------------
__global__ __launch_bounds__(256) void vq_main(const float* __restrict__ z,
                                               const unsigned short* __restrict__ embBf,
                                               const float* __restrict__ norms,
                                               int* __restrict__ idxOut) {
    __shared__ __align__(16) unsigned short btile[2][CTILE * LDS_STR];

    const int tid  = threadIdx.x;
    const int lane = tid & 31;
    const int wave = tid >> 5;
    const int half = lane >> 4;
    const int l15  = lane & 15;
    const int rowBase = blockIdx.x * MTILE + wave * 16;
    const int myRow   = rowBase + l15;

    // Kick off DMA of tile 0 before touching z (overlaps with A-frag loads)
    if (wave == 0) {
        tdm_load_tile(embBf, lds_addr_of(&btile[0][0]));
    }

    // --- this wave's 16 z-rows as 16 register A-fragments (bf16) ---
    // A 16x32 bf16 layout: lane(l15)=row M; elems 0..7 -> K = half*8 + e,
    //                      elems 8..15 -> K = 16 + half*8 + (e-8)
    v16bf afrag[16];
    const float* zr = z + (size_t)myRow * DIM;
#pragma unroll
    for (int kc = 0; kc < 16; ++kc) {
        const float* p0 = zr + kc * 32 + half * 8;
        const float* p1 = p0 + 16;
        float4 a0 = *(const float4*)(p0);
        float4 a1 = *(const float4*)(p0 + 4);
        float4 b0 = *(const float4*)(p1);
        float4 b1 = *(const float4*)(p1 + 4);
        v16bf f;
        f[0]  = cvt_bf16(a0.x); f[1]  = cvt_bf16(a0.y);
        f[2]  = cvt_bf16(a0.z); f[3]  = cvt_bf16(a0.w);
        f[4]  = cvt_bf16(a1.x); f[5]  = cvt_bf16(a1.y);
        f[6]  = cvt_bf16(a1.z); f[7]  = cvt_bf16(a1.w);
        f[8]  = cvt_bf16(b0.x); f[9]  = cvt_bf16(b0.y);
        f[10] = cvt_bf16(b0.z); f[11] = cvt_bf16(b0.w);
        f[12] = cvt_bf16(b1.x); f[13] = cvt_bf16(b1.y);
        f[14] = cvt_bf16(b1.z); f[15] = cvt_bf16(b1.w);
        afrag[kc] = f;
    }

    float bestD[8];
    int   bestI[8];
#pragma unroll
    for (int j = 0; j < 8; ++j) { bestD[j] = 3.4e38f; bestI[j] = 0; }

    for (int ct = 0; ct < NTILES; ++ct) {
        const int cur = ct & 1;
        // Wave 0: tile ct has landed when TENSORcnt drains to 0.
        if (wave == 0) {
            __builtin_amdgcn_s_wait_tensorcnt(0);
        }
        // One barrier: publishes btile[cur] (tile ct) to all waves AND retires
        // btile[1-cur] (everyone finished computing tile ct-1 to get here).
        __syncthreads();
        // Immediately stream the next tile into the retired buffer.
        if (wave == 0 && (ct + 1) < NTILES) {
            tdm_load_tile(embBf + (size_t)(ct + 1) * CTILE * DIM,
                          lds_addr_of(&btile[1 - cur][0]));
        }

#pragma unroll
        for (int sub = 0; sub < 2; ++sub) {
            v8f acc = {0.f, 0.f, 0.f, 0.f, 0.f, 0.f, 0.f, 0.f};
            // B 32x16 bf16 layout: lane(l15)=col N=code; elems 0..7 -> K = half*8 + e, elems 8..15 -> K = 16 + half*8 + (e-8)
            const unsigned short* bp =
                &btile[cur][(sub * 16 + l15) * LDS_STR + half * 8];
#pragma unroll
            for (int kc = 0; kc < 16; ++kc) {
                union { uint4 u[2]; v16bf v; } bb;
                bb.u[0] = *(const uint4*)(bp + kc * 32);
                bb.u[1] = *(const uint4*)(bp + kc * 32 + 16);
                acc = __builtin_amdgcn_wmma_f32_16x16x32_bf16(
                    false, afrag[kc], false, bb.v, (short)0, acc, false, false);
            }
            const int   code = ct * CTILE + sub * 16 + l15;
            const float nrm  = norms[code];
            // C layout: elem j -> (M = j + 8*half, N = l15)
#pragma unroll
            for (int j = 0; j < 8; ++j) {
                float d = nrm - 2.0f * acc[j];
                if (d < bestD[j]) { bestD[j] = d; bestI[j] = code; }
            }
        }
    }

    // argmin across the 16 lanes (N dimension) sharing each row half
#pragma unroll
    for (int j = 0; j < 8; ++j) {
        float d  = bestD[j];
        int   ix = bestI[j];
#pragma unroll
        for (int m = 1; m < 16; m <<= 1) {
            float od = __shfl_xor(d, m, 32);
            int   oi = __shfl_xor(ix, m, 32);
            if (od < d || (od == d && oi < ix)) { d = od; ix = oi; }
        }
        if (l15 == 0) {
            int row = rowBase + j + 8 * half;
            idxOut[row] = ix;
        }
    }
}

// ---------------------------------------------------------------------------
// Kernel 3: gather quantized rows (f32) + per-row squared-error partial
// ---------------------------------------------------------------------------
__global__ __launch_bounds__(256) void vq_gather(const float* __restrict__ z,
                                                 const float* __restrict__ emb,
                                                 const int* __restrict__ idx,
                                                 float* __restrict__ qOut,
                                                 float* __restrict__ partials) {
    const int row = blockIdx.x;
    const int t   = threadIdx.x;
    const int code = idx[row];
    const float* e  = emb + (size_t)code * DIM;
    const float* zr = z + (size_t)row * DIM;
    float* q = qOut + (size_t)row * DIM;
    float s = 0.f;
    for (int i = t; i < DIM; i += 256) {
        float qv = e[i];
        q[i] = qv;
        float d = zr[i] - qv;
        s += d * d;
    }
    __shared__ float red[256];
    red[t] = s;
    __syncthreads();
    for (int off = 128; off > 0; off >>= 1) {
        if (t < off) red[t] += red[t + off];
        __syncthreads();
    }
    if (t == 0) partials[row] = red[0];
}

// ---------------------------------------------------------------------------
// Kernel 4: deterministic final reduction -> commitment scalar
// ---------------------------------------------------------------------------
__global__ __launch_bounds__(256) void vq_commit(const float* __restrict__ partials,
                                                 float* __restrict__ out) {
    float s = 0.f;
    for (int i = threadIdx.x; i < N_PTS; i += 256) s += partials[i];
    __shared__ float red[256];
    red[threadIdx.x] = s;
    __syncthreads();
    for (int off = 128; off > 0; off >>= 1) {
        if (threadIdx.x < off) red[threadIdx.x] += red[threadIdx.x + off];
        __syncthreads();
    }
    if (threadIdx.x == 0) out[0] = red[0] / (float)((size_t)N_PTS * DIM);
}

// ---------------------------------------------------------------------------
extern "C" void kernel_launch(void* const* d_in, const int* in_sizes, int n_in,
                              void* d_out, int out_size, void* d_ws, size_t ws_size,
                              hipStream_t stream) {
    (void)in_sizes; (void)n_in; (void)out_size; (void)ws_size;

    const float* z   = (const float*)d_in[0];
    const float* emb = (const float*)d_in[1];

    float* out       = (float*)d_out;
    float* qOut      = out;                                   // [N, DIM] f32
    int*   idxOut    = (int*)(out + (size_t)N_PTS * DIM);     // [N] int32
    float* commitOut = out + (size_t)N_PTS * DIM + N_PTS;     // [1] f32

    unsigned short* embBf = (unsigned short*)d_ws;            // 8 MB bf16 emb
    float* norms    = (float*)((char*)d_ws + (size_t)NUM_CODES * DIM * 2);
    float* partials = norms + NUM_CODES;                      // [N] f32

    vq_prep<<<NUM_CODES, 256, 0, stream>>>(emb, embBf, norms);
    vq_main<<<N_PTS / MTILE, 256, 0, stream>>>(z, embBf, norms, idxOut);
    vq_gather<<<N_PTS, 256, 0, stream>>>(z, emb, idxOut, qOut, partials);
    vq_commit<<<1, 256, 0, stream>>>(partials, commitOut);
}
